// VisionTransformer_20091857010969
// MI455X (gfx1250) — compile-verified, matrix operand order checked by reading, not run on hardware
//
#include <hip/hip_runtime.h>
#include <hip/hip_bf16.h>

typedef __attribute__((ext_vector_type(16))) _Float16 v16h;
typedef __attribute__((ext_vector_type(8)))  float    v8f;

namespace {
constexpr int   kB = 16, kC = 3, kH = 128, kW = 128;
constexpr int   kE = 512, kNH = 8, kP = 64, kHID = 2048;
constexpr int   kS = 257, kPW = 16;
constexpr float kScale = 0.04419417382415922f;   // 1/sqrt(512)

constexpr long ZSZ = (long)kB * kS * kE;         // 2,105,344 floats
constexpr long ASZ = (long)kB * kNH * kS * kS;   // 8,454,272 floats
constexpr long PSZ = (long)kB * 256 * 192;       //   786,432 floats
constexpr long CSZ = (long)kB * kE;              //     8,192 floats
} // namespace

// ---------------------------------------------------------------------------
// Batched tiled GEMM on v_wmma_f32_16x16x32_f16.
// One wave owns a 32x32 output tile: 2 A-fragments x 2 B-fragments -> 4 WMMAs
// per 32-wide K step.  Hot loop is branch-free: OOB rows/cols are clamped
// (their results are computed but never stored), and the K tail is handled by
// clamped loads + zero selects.  fp32 in/out, f16 MACs, fp32 accumulate.
// BT=1 reads B transposed (B[n,k] row-major).
// ---------------------------------------------------------------------------
template <int BT>
__global__ __launch_bounds__(128) void gemm_wmma_kernel(
    const float* __restrict__ A, const float* __restrict__ Bm,
    float* __restrict__ Cm, const float* __restrict__ bias,
    int M, int N, int K, int lda, int ldb, int ldc,
    long sao, long sbo, long sco,       // outer-batch strides (A,B,C)
    long sai, long sbi, long sci,       // inner-batch strides (A,B,C)
    int inner, int act)
{
    const int wave = threadIdx.x >> 5;
    const int lane = threadIdx.x & 31;
    const int half = lane >> 4;          // ISA fragment half (lanes 16-31)
    const int loc  = lane & 15;
    const int n0 = (blockIdx.x * 4 + wave) * 32;
    const int m0 = blockIdx.y * 32;
    if (n0 >= N || m0 >= M) return;      // wave-uniform exit

    const int bo = blockIdx.z / inner;
    const int bi = blockIdx.z % inner;
    A  += bo * sao + bi * sai;
    Bm += bo * sbo + bi * sbi;
    Cm += bo * sco + bi * sci;

    // Clamped operand rows/cols: loads are unconditional; clamped lanes feed
    // only output rows/cols that the epilogue never stores.
    const float* Ar0 = A + (long)min(m0 + loc,      M - 1) * lda;
    const float* Ar1 = A + (long)min(m0 + 16 + loc, M - 1) * lda;
    const int cn0c = min(n0 + loc,      N - 1);
    const int cn1c = min(n0 + 16 + loc, N - 1);
    const float* Bc0 = BT ? (Bm + (long)cn0c * ldb) : (Bm + cn0c);
    const float* Bc1 = BT ? (Bm + (long)cn1c * ldb) : (Bm + cn1c);
    const long bstep = BT ? 1L : (long)ldb;     // B stride per K element

    v8f acc00 = {}, acc01 = {}, acc10 = {}, acc11 = {};
    const int kh  = half * 8;
    const int K32 = K & ~31;

    int k0 = 0;
    for (; k0 < K32; k0 += 32) {        // branch-free main loop
        v16h a0, a1, b0, b1;
#pragma unroll
        for (int e = 0; e < 8; ++e) {   // two contiguous 8-float runs per lane
            const int klo = k0 + kh + e, khi = k0 + 16 + kh + e;
            a0[e]     = (_Float16)Ar0[klo];
            a0[8 + e] = (_Float16)Ar0[khi];
            a1[e]     = (_Float16)Ar1[klo];
            a1[8 + e] = (_Float16)Ar1[khi];
            b0[e]     = (_Float16)Bc0[(long)klo * bstep];
            b0[8 + e] = (_Float16)Bc0[(long)khi * bstep];
            b1[e]     = (_Float16)Bc1[(long)klo * bstep];
            b1[8 + e] = (_Float16)Bc1[(long)khi * bstep];
        }
        if (k0 + 32 < K) {              // hint next K tile -> global_prefetch_b8
            __builtin_prefetch(&Ar0[k0 + 32 + kh], 0, 1);
            __builtin_prefetch(&Bc0[(long)(k0 + 32 + kh) * bstep], 0, 1);
        }
        acc00 = __builtin_amdgcn_wmma_f32_16x16x32_f16(false, a0, false, b0, (short)0, acc00, false, false);
        acc01 = __builtin_amdgcn_wmma_f32_16x16x32_f16(false, a0, false, b1, (short)0, acc01, false, false);
        acc10 = __builtin_amdgcn_wmma_f32_16x16x32_f16(false, a1, false, b0, (short)0, acc10, false, false);
        acc11 = __builtin_amdgcn_wmma_f32_16x16x32_f16(false, a1, false, b1, (short)0, acc11, false, false);
    }
    if (k0 < K) {                       // K tail: clamped loads + zero selects
        v16h a0, a1, b0, b1;
#pragma unroll
        for (int e = 0; e < 8; ++e) {
            const int klo = k0 + kh + e, khi = k0 + 16 + kh + e;
            const int kl = min(klo, K - 1), kk = min(khi, K - 1);
            const float va0 = Ar0[kl], wa0 = Ar0[kk];
            const float va1 = Ar1[kl], wa1 = Ar1[kk];
            const float vb0 = Bc0[(long)kl * bstep], wb0 = Bc0[(long)kk * bstep];
            const float vb1 = Bc1[(long)kl * bstep], wb1 = Bc1[(long)kk * bstep];
            a0[e]     = (_Float16)(klo < K ? va0 : 0.f);
            a0[8 + e] = (_Float16)(khi < K ? wa0 : 0.f);
            a1[e]     = (_Float16)(klo < K ? va1 : 0.f);
            a1[8 + e] = (_Float16)(khi < K ? wa1 : 0.f);
            b0[e]     = (_Float16)(klo < K ? vb0 : 0.f);
            b0[8 + e] = (_Float16)(khi < K ? wb0 : 0.f);
            b1[e]     = (_Float16)(klo < K ? vb1 : 0.f);
            b1[8 + e] = (_Float16)(khi < K ? wb1 : 0.f);
        }
        acc00 = __builtin_amdgcn_wmma_f32_16x16x32_f16(false, a0, false, b0, (short)0, acc00, false, false);
        acc01 = __builtin_amdgcn_wmma_f32_16x16x32_f16(false, a0, false, b1, (short)0, acc01, false, false);
        acc10 = __builtin_amdgcn_wmma_f32_16x16x32_f16(false, a1, false, b0, (short)0, acc10, false, false);
        acc11 = __builtin_amdgcn_wmma_f32_16x16x32_f16(false, a1, false, b1, (short)0, acc11, false, false);
    }

    // Epilogue: C/D fragment layout row = r + 8*half, col = lane&15.
    const int c0 = n0 + loc, c1 = n0 + 16 + loc;
    auto emit = [&](float v, int rm, int cn) {
        if (rm < M && cn < N) {
            if (bias) v += bias[cn];
            if (act)  v = v > 0.f ? v : 0.2f * v;
            Cm[(long)rm * ldc + cn] = v;
        }
    };
#pragma unroll
    for (int r = 0; r < 8; ++r) {
        const int rm0 = m0 + r + half * 8;
        const int rm1 = rm0 + 16;
        emit(acc00[r], rm0, c0);
        emit(acc01[r], rm0, c1);
        emit(acc10[r], rm1, c0);
        emit(acc11[r], rm1, c1);
    }
}

// --------------------------- wave-per-row LayerNorm ------------------------
__global__ __launch_bounds__(128) void ln_kernel(
    float* __restrict__ out, const float* __restrict__ xa,
    const float* __restrict__ xb, const float* __restrict__ g,
    const float* __restrict__ b, int rows, int E)
{
    const int row  = blockIdx.x * 4 + (threadIdx.x >> 5);
    const int lane = threadIdx.x & 31;
    if (row >= rows) return;
    const float* pa = xa + (long)row * E;
    const float* pb = xb ? xb + (long)row * E : nullptr;
    float s = 0.f, s2 = 0.f;
    for (int i = lane; i < E; i += 32) {
        float v = pa[i] + (pb ? pb[i] : 0.f);
        s += v; s2 += v * v;
    }
    for (int off = 16; off; off >>= 1) {
        s  += __shfl_xor(s,  off, 32);
        s2 += __shfl_xor(s2, off, 32);
    }
    const float mean = s / E;
    const float inv  = rsqrtf(s2 / E - mean * mean + 1e-5f);
    float* po = out + (long)row * E;
    for (int i = lane; i < E; i += 32) {
        float v = pa[i] + (pb ? pb[i] : 0.f);
        po[i] = (v - mean) * inv * g[i] + b[i];
    }
}

// --------------------- wave-per-row scaled softmax (len n) -----------------
__global__ __launch_bounds__(128) void softmax_kernel(
    float* __restrict__ a, int rows, int n, float scale)
{
    const int row  = blockIdx.x * 4 + (threadIdx.x >> 5);
    const int lane = threadIdx.x & 31;
    if (row >= rows) return;
    float* p = a + (long)row * n;
    float mx = -3.4e38f;
    for (int i = lane; i < n; i += 32) mx = fmaxf(mx, p[i] * scale);
    for (int off = 16; off; off >>= 1) mx = fmaxf(mx, __shfl_xor(mx, off, 32));
    float sum = 0.f;
    for (int i = lane; i < n; i += 32) {
        float e = __expf(p[i] * scale - mx);
        p[i] = e; sum += e;
    }
    for (int off = 16; off; off >>= 1) sum += __shfl_xor(sum, off, 32);
    const float inv = 1.f / sum;
    for (int i = lane; i < n; i += 32) p[i] *= inv;
}

// ------------- im2col for the 8x8/stride-8 patch conv (and inverse) --------
__global__ void patch_gather_kernel(const float* __restrict__ x,
                                    float* __restrict__ pa, long total)
{
    long idx = (long)blockIdx.x * blockDim.x + threadIdx.x;
    if (idx >= total) return;
    const int n  = idx % 192;
    const int hw = (idx / 192) % 256;
    const int b  = idx / (192 * 256);
    const int c = n / 64, rp = (n % 64) / 8, q = n % 8;
    const int h = hw / kPW, w = hw % kPW;
    pa[idx] = x[(((long)b * kC + c) * kH + h * 8 + rp) * kW + w * 8 + q];
}

__global__ void deconv_scatter_kernel(const float* __restrict__ g,
                                      const float* __restrict__ bias,
                                      float* __restrict__ out, long total)
{
    long idx = (long)blockIdx.x * blockDim.x + threadIdx.x;
    if (idx >= total) return;
    const int n  = idx % 192;
    const int hw = (idx / 192) % 256;
    const int b  = idx / (192 * 256);
    const int c = n / 64, rp = (n % 64) / 8, q = n % 8;
    const int h = hw / kPW, w = hw % kPW;
    out[(((long)b * kC + c) * kH + h * 8 + rp) * kW + w * 8 + q] = g[idx] + bias[c];
}

// ------------------- cond MLP layer 1 (K=3: scalar kernel) -----------------
__global__ void cond1_kernel(const float* __restrict__ y,
                             const float* __restrict__ w1,
                             const float* __restrict__ b1,
                             float* __restrict__ out, long total)
{
    long idx = (long)blockIdx.x * blockDim.x + threadIdx.x;
    if (idx >= total) return;
    const int e = idx % kE, b = idx / kE;
    float s = b1[e];
    for (int j = 0; j < 3; ++j) s += y[b * 3 + j] * w1[j * kE + e];
    out[idx] = s > 0.f ? s : 0.2f * s;
}

// ------------ build z = concat(cond, tok) + pos_embed ----------------------
__global__ void concat_kernel(const float* __restrict__ cond,
                              const float* __restrict__ tok,
                              const float* __restrict__ pos,
                              float* __restrict__ z, long total)
{
    long idx = (long)blockIdx.x * blockDim.x + threadIdx.x;
    if (idx >= total) return;
    const int e = idx % kE;
    const int s = (idx / kE) % kS;
    const int b = idx / ((long)kE * kS);
    float v = (s == 0) ? cond[(long)b * kE + e]
                       : tok[((long)b * 256 + s - 1) * kE + e];
    z[idx] = v + pos[(long)s * kE + e];
}

// ------------- (B,NH,S,P) -> (B,S,NH*P) head merge -------------------------
__global__ void merge_heads_kernel(const float* __restrict__ o,
                                   float* __restrict__ o2, long total)
{
    long idx = (long)blockIdx.x * blockDim.x + threadIdx.x;
    if (idx >= total) return;
    const int e = idx % kE;
    const int s = (idx / kE) % kS;
    const int b = idx / ((long)kE * kS);
    const int h = e / kP, p = e % kP;
    o2[idx] = o[(((long)b * kNH + h) * kS + s) * (long)kP + p];
}

// ---------------------------------------------------------------------------
static void gemm(hipStream_t st, const float* A, const float* Bm, float* Cm,
                 const float* bias, int M, int N, int K,
                 int lda, int ldb, int ldc,
                 long sao, long sbo, long sco, long sai, long sbi, long sci,
                 int outer, int inner, int btrans, int act)
{
    dim3 grid((N + 127) / 128, (M + 31) / 32, outer * inner);
    if (btrans)
        gemm_wmma_kernel<1><<<grid, 128, 0, st>>>(A, Bm, Cm, bias, M, N, K,
            lda, ldb, ldc, sao, sbo, sco, sai, sbi, sci, inner, act);
    else
        gemm_wmma_kernel<0><<<grid, 128, 0, st>>>(A, Bm, Cm, bias, M, N, K,
            lda, ldb, ldc, sao, sbo, sco, sai, sbi, sci, inner, act);
}

static inline dim3 ew_grid(long n) { return dim3((unsigned)((n + 255) / 256)); }

extern "C" void kernel_launch(void* const* d_in, const int* in_sizes, int n_in,
                              void* d_out, int out_size, void* d_ws, size_t ws_size,
                              hipStream_t stream)
{
    (void)in_sizes; (void)n_in; (void)out_size; (void)ws_size;
    const float* x        = (const float*)d_in[0];
    const float* y        = (const float*)d_in[1];
    const float* patch_w  = (const float*)d_in[2];
    const float* patch_b  = (const float*)d_in[3];
    const float* cond_w1  = (const float*)d_in[4];
    const float* cond_b1  = (const float*)d_in[5];
    const float* cond_w2  = (const float*)d_in[6];
    const float* cond_b2  = (const float*)d_in[7];
    const float* pos      = (const float*)d_in[8];
    const float* final_g  = (const float*)d_in[33];
    const float* final_b  = (const float*)d_in[34];
    const float* deconv_w = (const float*)d_in[35];
    const float* deconv_b = (const float*)d_in[36];
    float* out = (float*)d_out;

    float* ws   = (float*)d_ws;
    float* z    = ws;                 // (B,S,E)   persistent activations
    float* t1   = ws + ZSZ;           // k / o2 / h2 / zn
    float* t2   = ws + 2 * ZSZ;       // v / mhsa-out
    float* t3   = ws + 3 * ZSZ;       // qv / o / z2
    float* big  = ws + 4 * ZSZ;       // attn matrix (B,NH,S,S)  OR  FF hidden
    float* pat  = big + ASZ;          // patch matrix / deconv grid
    float* c1   = pat + PSZ;          // cond hidden
    float* c2   = c1 + CSZ;           // cond token

    const int  rowsBS = kB * kS;
    const long sSP = (long)kS * kP, sSS = (long)kS * kS;

    // ---- patch embedding: im2col + GEMM (B^T), bias fused -------------------
    patch_gather_kernel<<<ew_grid(PSZ), 256, 0, stream>>>(x, pat, PSZ);
    gemm(stream, pat, patch_w, t1, patch_b, kB * 256, kE, 192,
         192, 192, kE, 0, 0, 0, 0, 0, 0, 1, 1, /*btrans=*/1, 0);
    // ---- cond MLP -----------------------------------------------------------
    cond1_kernel<<<ew_grid(CSZ), 256, 0, stream>>>(y, cond_w1, cond_b1, c1, CSZ);
    gemm(stream, c1, cond_w2, c2, cond_b2, kB, kE, kE,
         kE, kE, kE, 0, 0, 0, 0, 0, 0, 1, 1, 0, /*leaky*/1);
    // ---- z = concat + pos ---------------------------------------------------
    concat_kernel<<<ew_grid(ZSZ), 256, 0, stream>>>(c2, t1, pos, z, ZSZ);

    // ---- 12 transformer blocks ----------------------------------------------
    for (int i = 0; i < 12; ++i) {
        const bool enc = i < 6;
        const int  li  = enc ? i : i - 6;
        const float* vp  = (const float*)d_in[enc ?  9 : 21] + (long)li * kNH * kE * kP;
        const float* kp  = (const float*)d_in[enc ? 10 : 22] + (long)li * kNH * kE * kP;
        const float* qh  = (const float*)d_in[enc ? 11 : 23] + (long)li * kNH * kS * kS;
        const float* lf  = (const float*)d_in[enc ? 12 : 24] + (long)li * kNH * kP * kE;
        const float* g1  = (const float*)d_in[enc ? 13 : 25] + (long)li * kE;
        const float* b1  = (const float*)d_in[enc ? 14 : 26] + (long)li * kE;
        const float* w1  = (const float*)d_in[enc ? 15 : 27] + (long)li * kE * kHID;
        const float* fb1 = (const float*)d_in[enc ? 16 : 28] + (long)li * kHID;
        const float* w2  = (const float*)d_in[enc ? 17 : 29] + (long)li * kHID * kE;
        const float* fb2 = (const float*)d_in[enc ? 18 : 30] + (long)li * kE;
        const float* g2  = (const float*)d_in[enc ? 19 : 31] + (long)li * kE;
        const float* b2  = (const float*)d_in[enc ? 20 : 32] + (long)li * kE;

        // k = x@kp, v = x@vp   (outer=head, inner=batch; A shared across heads)
        gemm(stream, z, kp, t1, nullptr, kS, kP, kE, kE, kP, kP,
             0, (long)kE * kP, sSP, (long)kS * kE, 0, (long)kNH * sSP,
             kNH, kB, 0, 0);
        gemm(stream, z, vp, t2, nullptr, kS, kP, kE, kE, kP, kP,
             0, (long)kE * kP, sSP, (long)kS * kE, 0, (long)kNH * sSP,
             kNH, kB, 0, 0);
        // qv = qh @ v   (qh shared across batch)
        gemm(stream, qh, t2, t3, nullptr, kS, kP, kS, kS, kP, kP,
             sSS, sSP, sSP, 0, (long)kNH * sSP, (long)kNH * sSP,
             kNH, kB, 0, 0);
        // a[t,s] = k[t,:] . qv[s,:]   (B transposed)
        gemm(stream, t1, t3, big, nullptr, kS, kS, kP, kP, kP, kS,
             sSP, sSP, sSS, (long)kNH * sSP, (long)kNH * sSP, (long)kNH * sSS,
             kNH, kB, /*btrans=*/1, 0);
        // softmax over last dim with 1/sqrt(E) scaling
        softmax_kernel<<<dim3((kB * kNH * kS + 3) / 4), 128, 0, stream>>>(
            big, kB * kNH * kS, kS, kScale);
        // o = a @ v
        gemm(stream, big, t2, t3, nullptr, kS, kP, kS, kS, kP, kP,
             sSS, sSP, sSP, (long)kNH * sSS, (long)kNH * sSP, (long)kNH * sSP,
             kNH, kB, 0, 0);
        // (B,NH,S,P) -> (B,S,E), then lift
        merge_heads_kernel<<<ew_grid(ZSZ), 256, 0, stream>>>(t3, t1, ZSZ);
        gemm(stream, t1, lf, t2, nullptr, rowsBS, kE, kE, kE, kE, kE,
             0, 0, 0, 0, 0, 0, 1, 1, 0, 0);
        // z2 = LN(z + mhsa)
        ln_kernel<<<dim3((rowsBS + 3) / 4), 128, 0, stream>>>(
            t3, z, t2, g1, b1, rowsBS, kE);
        // FFN: h1 = leaky(z2@w1 + b1);  h2 = h1@w2 + b2
        gemm(stream, t3, w1, big, fb1, rowsBS, kHID, kE, kE, kHID, kHID,
             0, 0, 0, 0, 0, 0, 1, 1, 0, /*leaky*/1);
        gemm(stream, big, w2, t1, fb2, rowsBS, kE, kHID, kHID, kE, kE,
             0, 0, 0, 0, 0, 0, 1, 1, 0, 0);
        // z = LN(z2 + h)
        ln_kernel<<<dim3((rowsBS + 3) / 4), 128, 0, stream>>>(
            z, t3, t1, g2, b2, rowsBS, kE);
    }

    // ---- final LN (token 0 normalized too, then skipped via +E offset) ------
    ln_kernel<<<dim3((rowsBS + 3) / 4), 128, 0, stream>>>(
        t1, z, nullptr, final_g, final_b, rowsBS, kE);
    // ---- deconv: per-image GEMM (skip cond token), then scatter + bias ------
    gemm(stream, t1 + kE, deconv_w, pat, nullptr, 256, 192, kE,
         kE, 192, 192, 0, 0, 0, (long)kS * kE, 0, (long)256 * 192,
         1, kB, 0, 0);
    deconv_scatter_kernel<<<ew_grid(PSZ), 256, 0, stream>>>(pat, deconv_b, out, PSZ);
}
